// Drug_Attention_54786602828487
// MI455X (gfx1250) — matrix operand order repaired and checked by reading, not compiled, not run on hardware
//
#include <hip/hip_runtime.h>

typedef __bf16 bf16;
typedef bf16  v16bf __attribute__((ext_vector_type(16)));
typedef bf16  v8bf  __attribute__((ext_vector_type(8)));
typedef bf16  v4bf  __attribute__((ext_vector_type(4)));
typedef float v8f   __attribute__((ext_vector_type(8)));

#define NB   64            // sequence length N
#define DB   512           // feature dim D
#define QSCALE 0.04419417382415922f   // 1/sqrt(512)

__device__ __forceinline__ v8f v8f_zero() {
    v8f z = {0.f, 0.f, 0.f, 0.f, 0.f, 0.f, 0.f, 0.f};
    return z;
}

__device__ __forceinline__ v8f wmma_bf16(v16bf a, v16bf b, v8f c) {
    // D = A(16x32 bf16) * B(32x16 bf16) + C(16x16 f32)
    return __builtin_amdgcn_wmma_f32_16x16x32_bf16(
        /*neg_a=*/false, a, /*neg_b=*/false, b,
        /*c_mod=*/(short)0, c, /*reuse_a=*/false, /*reuse_b=*/false);
}

// A-fragment: 16x32 bf16 tile of row-major [rows][lda] matrix.
// lane<16:  m = row0+lane,    K = k0+0..15  (v[0..15])
// lane>=16: m = row0+lane-16, K = k0+16..31 (v[0..15])
__device__ __forceinline__ v16bf load_a_frag(const bf16* src, int lda,
                                             int row0, int k0, int lane) {
    const bf16* p = src + (row0 + (lane & 15)) * lda + k0 + ((lane >> 4) << 4);
    v8bf lo = *reinterpret_cast<const v8bf*>(p);        // 16B aligned
    v8bf hi = *reinterpret_cast<const v8bf*>(p + 8);    // 16B aligned
    return __builtin_shufflevector(lo, hi, 0, 1, 2, 3, 4, 5, 6, 7,
                                   8, 9, 10, 11, 12, 13, 14, 15);
}

// B-fragment (32x16, B[k][n] = M[n0+n][k0+k] for row-major M): lane = N column,
// each lane reads 16 consecutive K elements of row (n0+lane%16), half by lane/16.
// f32-weight variant (fallback path; converts in-flight):
__device__ __forceinline__ v16bf load_b_frag_w(const float* W, int ldw,
                                               int n0, int k0, int lane) {
    const float* p = W + (size_t)(n0 + (lane & 15)) * ldw + k0 + ((lane >> 4) << 4);
    v16bf b;
#pragma unroll
    for (int j = 0; j < 16; ++j) b[j] = (bf16)p[j];
    return b;
}
// bf16-weight variant (primary path; weights pre-converted into d_ws):
__device__ __forceinline__ v16bf load_b_frag_w(const bf16* W, int ldw,
                                               int n0, int k0, int lane) {
    const bf16* p = W + (size_t)(n0 + (lane & 15)) * ldw + k0 + ((lane >> 4) << 4);
    return *reinterpret_cast<const v16bf*>(p);          // 32B aligned
}

__device__ __forceinline__ v16bf load_b_frag_lds(const bf16* src, int lds_,
                                                 int n0, int k0, int lane) {
    const bf16* p = src + (n0 + (lane & 15)) * lds_ + k0 + ((lane >> 4) << 4);
    return *reinterpret_cast<const v16bf*>(p);          // 32B aligned
}

// C-tile (16x16 f32) store to bf16 row-major [ld], scaled.
__device__ __forceinline__ void store_c_bf16(bf16* dst, int ld, int row0, int col0,
                                             int lane, v8f c, float scale) {
    int n  = col0 + (lane & 15);
    int mb = row0 + ((lane >> 4) << 3);
#pragma unroll
    for (int j = 0; j < 8; ++j) dst[(mb + j) * ld + n] = (bf16)(c[j] * scale);
}

// Transposed C-tile store: dst[col][row] (contiguous 16B per lane).
__device__ __forceinline__ void store_c_bf16_t(bf16* dst, int ld, int row0, int col0,
                                               int lane, v8f c) {
    int n  = col0 + (lane & 15);
    int mb = row0 + ((lane >> 4) << 3);
    bf16* p = dst + n * ld + mb;
#pragma unroll
    for (int j = 0; j < 8; ++j) p[j] = (bf16)c[j];
}

// out[64][512] = xs[64][512] * W^T ; wave owns 4 column tiles, all 4 row tiles,
// so the workgroup streams each weight matrix exactly once (L2-resident).
template <bool TRANS, typename WT>
__device__ __forceinline__ void projection(const bf16* xs, const WT* W,
                                           bf16* dst, int wave, int lane,
                                           float scale) {
#pragma unroll 1
    for (int c = 0; c < 4; ++c) {
        const int n0 = (wave * 4 + c) * 16;
        v8f acc[4];
#pragma unroll
        for (int r = 0; r < 4; ++r) acc[r] = v8f_zero();
#pragma unroll 1
        for (int k0 = 0; k0 < DB; k0 += 32) {
            v16bf bf = load_b_frag_w(W, DB, n0, k0, lane);
#pragma unroll
            for (int r = 0; r < 4; ++r) {
                v16bf af = load_a_frag(xs, DB, r * 16, k0, lane);
                acc[r] = wmma_bf16(af, bf, acc[r]);
            }
        }
#pragma unroll
        for (int r = 0; r < 4; ++r) {
            if (TRANS) store_c_bf16_t(dst, NB, r * 16, n0, lane, acc[r]);
            else       store_c_bf16(dst, DB, r * 16, n0, lane, acc[r], scale);
        }
    }
}

// ---- prep: convert Wq|Wk|Wv (f32, 512x512 each) -> contiguous bf16 in ws ----
__global__ __launch_bounds__(256)
void convert_w_kernel(const float* __restrict__ Wq, const float* __restrict__ Wk,
                      const float* __restrict__ Wv, bf16* __restrict__ dst) {
    const int i = blockIdx.x * 256 + threadIdx.x;   // float4 index, 3*65536 total
    const int m = i >> 16;                          // which matrix
    const int j = i & 65535;                        // float4 within matrix
    const float* W = (m == 0) ? Wq : ((m == 1) ? Wk : Wv);
    float4 v = reinterpret_cast<const float4*>(W)[j];
    v4bf o = {(bf16)v.x, (bf16)v.y, (bf16)v.z, (bf16)v.w};
    *reinterpret_cast<v4bf*>(dst + (size_t)i * 4) = o;   // 8B store
}

template <typename WT>
__global__ __launch_bounds__(256)
void drug_attn_fused_kernel(const float* __restrict__ x,
                            const WT* __restrict__ Wq,
                            const WT* __restrict__ Wk,
                            const WT* __restrict__ Wv,
                            float* __restrict__ out) {
    extern __shared__ char smem_raw[];
    bf16* xs = reinterpret_cast<bf16*>(smem_raw);   // 64x512 bf16 : 64 KB
    bf16* Qs = xs + NB * DB;                        // 64x512 bf16 : 64 KB
    bf16* Ks = Qs + NB * DB;                        // 64x512 bf16 : 64 KB
    bf16* Vt = Ks + NB * DB;                        // 512x64 bf16 : 64 KB (V^T)
    // Overlays inside the xs region (xs is dead after projections):
    float* S = reinterpret_cast<float*>(smem_raw);               // 64x64 f32
    bf16*  P = reinterpret_cast<bf16*>(smem_raw + NB * NB * 4);  // 64x64 bf16

    const int tid  = threadIdx.x;
    const int wave = tid >> 5;
    const int lane = tid & 31;
    const int bidx = blockIdx.x;

    // ---- Phase 1: stage x[b] -> LDS bf16 ---------------------------------
    const float4* xsrc = reinterpret_cast<const float4*>(x + (size_t)bidx * NB * DB);
#pragma unroll 1
    for (int i = tid; i < (NB * DB / 4); i += 256) {
        float4 v = xsrc[i];
        v4bf o = {(bf16)v.x, (bf16)v.y, (bf16)v.z, (bf16)v.w};
        *reinterpret_cast<v4bf*>(xs + (i << 2)) = o;
    }
    __syncthreads();

    // ---- Phase 2: Q,K,V projections --------------------------------------
    projection<false>(xs, Wq, Qs, wave, lane, QSCALE);  // fold 1/sqrt(D) into Q
    projection<false>(xs, Wk, Ks, wave, lane, 1.0f);
    projection<true >(xs, Wv, Vt, wave, lane, 1.0f);    // store V transposed
    __syncthreads();

    // ---- Phase 3: scores S = Q * K^T (16 tiles, 2 per wave) --------------
#pragma unroll 1
    for (int t = wave; t < 16; t += 8) {
        const int q0 = (t >> 2) * 16;   // query rows
        const int m0 = (t & 3) * 16;    // key cols
        v8f acc = v8f_zero();
#pragma unroll 1
        for (int k0 = 0; k0 < DB; k0 += 32) {
            v16bf af = load_a_frag(Qs, DB, q0, k0, lane);
            v16bf bf = load_b_frag_lds(Ks, DB, m0, k0, lane);
            acc = wmma_bf16(af, bf, acc);
        }
        const int n  = m0 + (lane & 15);
        const int mb = q0 + ((lane >> 4) << 3);
#pragma unroll
        for (int j = 0; j < 8; ++j) S[(mb + j) * NB + n] = acc[j];
    }
    __syncthreads();

    // ---- Phase 4: row softmax -> P (bf16) --------------------------------
    if (tid < NB) {
        float* row = S + tid * NB;
        float mx = row[0];
#pragma unroll 1
        for (int m = 1; m < NB; ++m) mx = fmaxf(mx, row[m]);
        float s = 0.f;
#pragma unroll 1
        for (int m = 0; m < NB; ++m) {
            float e = __expf(row[m] - mx);
            row[m] = e;
            s += e;
        }
        float inv = 1.0f / s;
        bf16* prow = P + tid * NB;
#pragma unroll 1
        for (int m = 0; m < NB; ++m) prow[m] = (bf16)(row[m] * inv);
    }
    __syncthreads();

    // ---- Phase 5: combined = P * V, fused column-sum over n, store -------
#pragma unroll 1
    for (int c = 0; c < 4; ++c) {
        const int v0 = (wave * 4 + c) * 16;   // output feature tile
        float colsum = 0.f;
#pragma unroll
        for (int r = 0; r < 4; ++r) {         // 4 query-row tiles
            v8f acc = v8f_zero();
#pragma unroll
            for (int k0 = 0; k0 < NB; k0 += 32) {
                v16bf af = load_a_frag(P, NB, r * 16, k0, lane);
                v16bf bf = load_b_frag_lds(Vt, NB, v0, k0, lane);
                acc = wmma_bf16(af, bf, acc);
            }
#pragma unroll
            for (int j = 0; j < 8; ++j) colsum += acc[j];
        }
        colsum += __shfl_xor(colsum, 16, 32);  // combine the two row-halves
        if (lane < 16) out[(size_t)bidx * DB + v0 + lane] = colsum;
    }
}

extern "C" void kernel_launch(void* const* d_in, const int* in_sizes, int n_in,
                              void* d_out, int out_size, void* d_ws, size_t ws_size,
                              hipStream_t stream) {
    (void)n_in; (void)out_size;
    const float* x  = (const float*)d_in[0];
    const float* Wq = (const float*)d_in[1];
    const float* Wk = (const float*)d_in[2];
    const float* Wv = (const float*)d_in[3];
    float* out = (float*)d_out;

    const int B = in_sizes[0] / (NB * DB);                        // 2048
    const size_t lds_bytes = (size_t)4 * NB * DB * sizeof(bf16);  // 256 KB
    const size_t wbytes = (size_t)3 * DB * DB * sizeof(bf16);     // 1.5 MB

    if (ws_size >= wbytes) {
        // Pre-convert the three weight matrices to bf16 once per launch.
        bf16* wbf = (bf16*)d_ws;
        convert_w_kernel<<<dim3(3 * DB * DB / 4 / 256), dim3(256), 0, stream>>>(
            Wq, Wk, Wv, wbf);
        (void)hipFuncSetAttribute((const void*)&drug_attn_fused_kernel<bf16>,
                                  hipFuncAttributeMaxDynamicSharedMemorySize,
                                  (int)lds_bytes);
        drug_attn_fused_kernel<bf16><<<dim3(B), dim3(256), lds_bytes, stream>>>(
            x, wbf, wbf + DB * DB, wbf + 2 * DB * DB, out);
    } else {
        // Fallback: convert weights in-flight from f32.
        (void)hipFuncSetAttribute((const void*)&drug_attn_fused_kernel<float>,
                                  hipFuncAttributeMaxDynamicSharedMemorySize,
                                  (int)lds_bytes);
        drug_attn_fused_kernel<float><<<dim3(B), dim3(256), lds_bytes, stream>>>(
            x, Wq, Wk, Wv, out);
    }
}
